// VectorQuantizer_90958817395013
// MI455X (gfx1250) — matrix operand order repaired and checked by reading, not compiled, not run on hardware
//
#include <hip/hip_runtime.h>
#include <hip/hip_bf16.h>

#define B_ 8
#define D_ 512
#define T_ 2048
#define K_ 8192
#define BT_ (B_ * T_)        // 16384 rows
#define NCHUNK 2             // code-dim split per 16-row tile
#define CODES_PER_CHUNK (K_ / NCHUNK)   // 4096

typedef __attribute__((ext_vector_type(16))) __bf16 v16bf;
typedef __attribute__((ext_vector_type(8)))  __bf16 v8bf;
typedef __attribute__((ext_vector_type(8)))  float  v8f;

__device__ __forceinline__ unsigned short f32_to_bf16_rne(float f) {
    unsigned u = __float_as_uint(f);
    u += 0x7FFFu + ((u >> 16) & 1u);   // round-to-nearest-even
    return (unsigned short)(u >> 16);
}

// Monotonic float -> u32 key (total order matching <, incl. negatives).
__device__ __forceinline__ unsigned fkey(float d) {
    unsigned u = __float_as_uint(d);
    return (u & 0x80000000u) ? ~u : (u | 0x80000000u);
}

// ---------------- Kernel 0: reset packed argmin array ------------------------
__global__ void vq_init_best(unsigned long long* __restrict__ best) {
    int i = blockIdx.x * blockDim.x + threadIdx.x;
    if (i < BT_) best[i] = 0xFFFFFFFFFFFFFFFFull;
}

// ---------------- Kernel 1: codebook -> bf16, and e2[k] = ||E_k||^2 ----------
__global__ void vq_prep_embed(const float* __restrict__ E,
                              unsigned short* __restrict__ Ebf,
                              float* __restrict__ e2) {
    __shared__ float red[128];
    int k = blockIdx.x;
    const float* row = E + (size_t)k * D_;
    unsigned short* orow = Ebf + (size_t)k * D_;
    float s = 0.f;
    for (int d = threadIdx.x; d < D_; d += 128) {
        float v = row[d];
        s += v * v;
        orow[d] = f32_to_bf16_rne(v);
    }
    red[threadIdx.x] = s;
    __syncthreads();
    for (int off = 64; off > 0; off >>= 1) {
        if ((int)threadIdx.x < off) red[threadIdx.x] += red[threadIdx.x + off];
        __syncthreads();
    }
    if (threadIdx.x == 0) e2[k] = red[0];
}

// ---------------- Kernel 2: latents [B,D,T] f32 -> Xbf [B*T, D] bf16 ---------
__global__ void vq_prep_lat(const float* __restrict__ L,
                            unsigned short* __restrict__ X) {
    __shared__ float tile[32][33];
    int b  = blockIdx.z;
    int t0 = blockIdx.x * 32;
    int d0 = blockIdx.y * 32;
    int tx = threadIdx.x, ty = threadIdx.y;
    const float* base = L + (size_t)b * D_ * T_;
#pragma unroll
    for (int i = 0; i < 4; ++i) {
        int d = d0 + ty + 8 * i;
        tile[ty + 8 * i][tx] = base[(size_t)d * T_ + t0 + tx];
    }
    __syncthreads();
    unsigned short* xb = X + (size_t)b * T_ * D_;
#pragma unroll
    for (int i = 0; i < 4; ++i) {
        int t = t0 + ty + 8 * i;
        xb[(size_t)t * D_ + d0 + tx] = f32_to_bf16_rne(tile[tx][ty + 8 * i]);
    }
}

// ---------------- Kernel 3: WMMA GEMM + per-row argmin -----------------------
// One wave32 owns a 16-row A tile (register-resident, 128 VGPRs -> stays below
// the 256-VGPR direct-address limit so the compiler keeps deep load pipelining)
// and scans a 4096-code chunk. Cross-chunk merge via packed u64 atomic-min
// (exact lowest-index tie-break). 1024 row-tiles x 2 chunks = 2048 waves.
__global__ void __launch_bounds__(256)
vq_argmin(const unsigned short* __restrict__ Xbf,
          const unsigned short* __restrict__ Ebf,
          const float* __restrict__ e2,
          unsigned long long* __restrict__ best) {
    int wave = threadIdx.x >> 5;
    int tid  = threadIdx.x & 31;
    int wid  = blockIdx.x * 8 + wave;       // 0 .. 2047
    int rowTile = wid >> 1;                 // 1024 tiles of 16 rows
    int chunk   = wid & (NCHUNK - 1);
    int m  = tid & 15;                      // A: row-in-tile / B: col-in-tile
    int hi = tid >> 4;                      // half-wave selector

    // 16-bit A 16x32 layout: lane m holds row; VGPR0-3 <-> K=hi*8+0..7,
    // VGPR4-7 <-> K=16+hi*8+0..7  => two contiguous 16B loads per K-step.
    int row = rowTile * 16 + m;
    const v8bf* arow = (const v8bf*)(Xbf + (size_t)row * D_);
    v16bf a[16];
#pragma unroll
    for (int s = 0; s < 16; ++s) {
        v8bf lo = arow[s * 4 + hi];          // K = s*32 + hi*8 .. +7
        v8bf hh = arow[s * 4 + 2 + hi];      // K = s*32 + 16 + hi*8 .. +7
        a[s] = __builtin_shufflevector(lo, hh,
               0, 1, 2, 3, 4, 5, 6, 7, 8, 9, 10, 11, 12, 13, 14, 15);
    }

    float bestd[8];
    int   bestk[8];
#pragma unroll
    for (int j = 0; j < 8; ++j) { bestd[j] = 3.4e38f; bestk[j] = 0; }

    int nt0 = chunk * (CODES_PER_CHUNK / 16);
    for (int it = 0; it < CODES_PER_CHUNK / 16; ++it) {
        int n = (nt0 + it) * 16 + m;
        // 16-bit B 32x16 layout: lane holds col n; K = hi*8+0..7, then 16+hi*8+0..7 (matches A).
        const v8bf* brow = (const v8bf*)(Ebf + (size_t)n * D_);
        v8f c = {};
#pragma unroll
        for (int s = 0; s < 16; ++s) {
            v8bf blo = brow[s * 4 + hi];
            v8bf bhi = brow[s * 4 + 2 + hi];
            v16bf bm = __builtin_shufflevector(blo, bhi,
                       0, 1, 2, 3, 4, 5, 6, 7, 8, 9, 10, 11, 12, 13, 14, 15);
            c = __builtin_amdgcn_wmma_f32_16x16x32_bf16(
                    false, a[s], false, bm, (short)0, c, false, false);
        }
        float e2n = e2[n];
#pragma unroll
        for (int j = 0; j < 8; ++j) {
            // d2 ~ e2[n] - 2<x,e_n>  (row-constant x^2 dropped: argmin-invariant)
            float d = fmaf(-2.0f, c[j], e2n);
            bool better = d < bestd[j];
            bestd[j] = better ? d : bestd[j];
            bestk[j] = better ? n : bestk[j];
        }
    }

    // Butterfly min-reduce across the 16 lanes of each half-wave
    // (xor masks 1,2,4,8 never cross a 16-lane half under wave32).
#pragma unroll
    for (int off = 1; off < 16; off <<= 1) {
#pragma unroll
        for (int j = 0; j < 8; ++j) {
            float od = __shfl_xor(bestd[j], off, 32);
            int   ok = __shfl_xor(bestk[j], off, 32);
            bool take = (od < bestd[j]) || (od == bestd[j] && ok < bestk[j]);
            bestd[j] = take ? od : bestd[j];
            bestk[j] = take ? ok : bestk[j];
        }
    }

    // C/D layout: lanes 0-15 VGPR j -> row j; lanes 16-31 VGPR j -> row 8+j.
    if (m == 0) {
#pragma unroll
        for (int j = 0; j < 8; ++j) {
            int r = rowTile * 16 + hi * 8 + j;
            unsigned long long p =
                ((unsigned long long)fkey(bestd[j]) << 32) | (unsigned)bestk[j];
            atomicMin(&best[r], p);
        }
    }
}

// ---------------- Kernel 4: gather fp32 codes, transpose to [B,D,T] ----------
__global__ void vq_gather(const float* __restrict__ E,
                          const unsigned long long* __restrict__ best,
                          float* __restrict__ out) {
    __shared__ float tile[32][33];
    int b  = blockIdx.z;
    int t0 = blockIdx.x * 32;
    int d0 = blockIdx.y * 32;
    int tx = threadIdx.x, ty = threadIdx.y;
#pragma unroll
    for (int i = 0; i < 4; ++i) {
        int t  = t0 + ty + 8 * i;
        int id = (int)(best[b * T_ + t] & 0xFFFFFFFFu);
        tile[ty + 8 * i][tx] = E[(size_t)id * D_ + d0 + tx];
    }
    __syncthreads();
    float* ob = out + (size_t)b * D_ * T_;
#pragma unroll
    for (int i = 0; i < 4; ++i) {
        int d = d0 + ty + 8 * i;
        ob[(size_t)d * T_ + t0 + tx] = tile[tx][ty + 8 * i];
    }
}

extern "C" void kernel_launch(void* const* d_in, const int* in_sizes, int n_in,
                              void* d_out, int out_size, void* d_ws, size_t ws_size,
                              hipStream_t stream) {
    (void)in_sizes; (void)n_in; (void)out_size; (void)ws_size;
    const float* latents   = (const float*)d_in[0];   // [B, D, T] f32
    const float* embedding = (const float*)d_in[1];   // [K, D]    f32
    float* out = (float*)d_out;                       // [B, D, T] f32

    char* ws = (char*)d_ws;
    unsigned short* Ebf = (unsigned short*)(ws);                              //  8 MB
    unsigned short* Xbf = (unsigned short*)(ws + (size_t)8  * 1024 * 1024);   // 16 MB
    float* e2           = (float*)(ws + (size_t)24 * 1024 * 1024);            // 32 KB
    unsigned long long* best =
        (unsigned long long*)(ws + (size_t)24 * 1024 * 1024 + 65536);         // 128 KB

    vq_init_best<<<BT_ / 256, 256, 0, stream>>>(best);
    vq_prep_embed<<<K_, 128, 0, stream>>>(embedding, Ebf, e2);
    vq_prep_lat<<<dim3(T_ / 32, D_ / 32, B_), dim3(32, 8), 0, stream>>>(latents, Xbf);
    // 1024 row-tiles x 2 code-chunks = 2048 waves = 256 blocks of 8 waves
    vq_argmin<<<(BT_ / 16) * NCHUNK / 8, 256, 0, stream>>>(Xbf, Ebf, e2, best);
    vq_gather<<<dim3(T_ / 32, D_ / 32, B_), dim3(32, 8), 0, stream>>>(embedding, best, out);
}
